// GraphFlashback_35570919145678
// MI455X (gfx1250) — matrix operand order repaired and checked by reading, not compiled, not run on hardware
//
#include <hip/hip_runtime.h>
#include <hip/hip_bf16.h>
#include <math.h>

#define PP 20000
#define UU 10000
#define HH 128
#define BB 64
#define LL 100

typedef __attribute__((ext_vector_type(16))) _Float16 v16h;
typedef __attribute__((ext_vector_type(8)))  float    v8f;
typedef __attribute__((ext_vector_type(4)))  unsigned int u32x4;

union Frag { v16h h; u32x4 u[2]; };

// ---------------- zero workspace region ----------------
__global__ void k_zero(float* __restrict__ p, int n) {
    int i = blockIdx.x * 256 + threadIdx.x;
    if (i < n) p[i] = 0.f;
}

// ---------------- f32 -> f16 convert ----------------
__global__ void k_cvt(const float* __restrict__ s, _Float16* __restrict__ d, int n) {
    int i = blockIdx.x * 256 + threadIdx.x;
    if (i < n) d[i] = (_Float16)s[i];
}

// ---------------- COO segment-sum: dst[row] += val * emb[col] ----------------
// one block (128 threads = H) per edge
__global__ void k_scatter(const int* __restrict__ row, const int* __restrict__ col,
                          const float* __restrict__ val, const float* __restrict__ emb,
                          float* __restrict__ dst) {
    int e = blockIdx.x;
    int h = threadIdx.x;
    int r = row[e], c = col[e];
    float v = val[e];
    atomicAdd(dst + (size_t)r * HH + h, v * emb[(size_t)c * HH + h]);
}

// ---------------- gather x_emb + uls = exp(-||user_pref - x_emb||) ----------------
// grid = L*B blocks (idx = l*B + b), 128 threads
__global__ void k_gather_uls(const int* __restrict__ POI, const int* __restrict__ uid,
                             const float* __restrict__ encw, const float* __restrict__ encwu,
                             float* __restrict__ x_emb, float* __restrict__ uls) {
    int idx = blockIdx.x;
    int l = idx / BB, b = idx % BB;
    int h = threadIdx.x;
    int p = POI[b * LL + l];                 // POI_id[b,l] (time-major x = POI.T)
    float xe = encw[(size_t)p * HH + h];
    x_emb[(size_t)idx * HH + h] = xe;
    float up = encwu[(size_t)uid[b] * HH + h];
    float d = up - xe;
    __shared__ float red[HH];
    red[h] = d * d;
    __syncthreads();
    for (int s = 64; s > 0; s >>= 1) { if (h < s) red[h] += red[h + s]; __syncthreads(); }
    if (h == 0) uls[idx] = expf(-sqrtf(red[0]));
}

// ---------------- X[m,n] = x_emb[m,:] . W_ih[n,:] + b_ih[n] + b_hh[n] ----------------
__global__ void k_rnn_input(const float* __restrict__ x_emb, const float* __restrict__ W_ih,
                            const float* __restrict__ b_ih, const float* __restrict__ b_hh,
                            float* __restrict__ X) {
    __shared__ float xr[HH];
    int m = blockIdx.x, n = threadIdx.x;
    xr[n] = x_emb[(size_t)m * HH + n];
    __syncthreads();
    const float* w = W_ih + (size_t)n * HH;
    float acc = 0.f;
#pragma unroll 4
    for (int k = 0; k < HH; k++) acc = fmaf(xr[k], w[k], acc);
    X[(size_t)m * HH + n] = acc + b_ih[n] + b_hh[n];
}

// ---------------- sequential RNN via WMMA: h = tanh(X[l] + h @ W_hh^T) ----------------
// single block, 1024 threads = 32 waves; each wave owns one 16x16 tile of [64,128]
__global__ void __launch_bounds__(1024) k_rnn(const float* __restrict__ X,
        const float* __restrict__ W_hh, const float* __restrict__ memory,
        const int* __restrict__ uid, float* __restrict__ out) {
    __shared__ _Float16 hsh[BB * HH];          // 16 KB current state, f16
    int tid = threadIdx.x;
    int lane = tid & 31, wave = tid >> 5;
    int lrow = lane & 15, lhi = lane >> 4;
    int m0 = (wave & 3) * 16;                  // batch-row tile (0..48)
    int n0 = (wave >> 2) * 16;                 // hidden-col tile (0..112)

    // B fragments: B[k][n] = W_hh[n][k]; lane = N, 16 contiguous K halves per frag
    Frag bf[4];
#pragma unroll
    for (int kc = 0; kc < 4; kc++) {
        const float* wp = W_hh + (size_t)(n0 + lrow) * HH + kc * 32 + lhi * 16;
#pragma unroll
        for (int t = 0; t < 16; t++) bf[kc].h[t] = (_Float16)wp[t];
    }
    // h0 = memory[user_id]
    for (int i = tid; i < BB * HH; i += 1024) {
        int b = i >> 7, k = i & 127;
        hsh[i] = (_Float16)memory[(size_t)uid[b] * HH + k];
    }
    __syncthreads();

    for (int l = 0; l < LL; l++) {
        // A fragments from LDS: lane = M(batch row), 16 contiguous K halves per frag
        Frag af[4];
#pragma unroll
        for (int kc = 0; kc < 4; kc++) {
            const _Float16* hp = &hsh[(m0 + lrow) * HH + kc * 32 + lhi * 16];
            af[kc].u[0] = *(const u32x4*)hp;
            af[kc].u[1] = *(const u32x4*)(hp + 8);
        }
        // C init = input projection (f32)
        v8f acc;
#pragma unroll
        for (int r = 0; r < 8; r++)
            acc[r] = X[(size_t)(l * BB + m0 + r + lhi * 8) * HH + n0 + lrow];
#pragma unroll
        for (int kc = 0; kc < 4; kc++)
            acc = __builtin_amdgcn_wmma_f32_16x16x32_f16(false, af[kc].h, false, bf[kc].h,
                                                         (short)0, acc, false, false);
        __syncthreads();   // everyone has consumed old h
#pragma unroll
        for (int r = 0; r < 8; r++) {
            int row = m0 + r + lhi * 8;        // batch index
            int col = n0 + lrow;               // hidden index
            float v = tanhf(acc[r]);
            out[(size_t)(l * BB + row) * HH + col] = v;
            hsh[row * HH + col] = (_Float16)v;
        }
        __syncthreads();
    }
}

// ---------------- flashback attention + pack f16 GEMM A rows ----------------
// grid = L*B blocks (idx = i*B + b), 128 threads
__global__ void k_flash(const int* __restrict__ ts, const float* __restrict__ lat,
                        const float* __restrict__ lon, const float* __restrict__ uls,
                        const float* __restrict__ rout, const float* __restrict__ user_emb,
                        const int* __restrict__ uid, _Float16* __restrict__ A16) {
    int idx = blockIdx.x;
    int i = idx >> 6, b = idx & 63;
    int tx = threadIdx.x;
    __shared__ float wsh[128];
    __shared__ float red[128];
    float wv = 0.f;
    if (tx <= i) {
        float dt  = (float)(ts[b * LL + i] - ts[b * LL + tx]);   // >= 0 (sorted)
        float dla = lat[b * LL + i] - lat[b * LL + tx];
        float dlo = lon[b * LL + i] - lon[b * LL + tx];
        float dsn = sqrtf(dla * dla + dlo * dlo);
        float a = (cosf(dt * 7.2722052166430399e-05f) + 1.f) * 0.5f
                  * expf(dt * (-0.1f / 86400.f));
        float e = expf(-1000.f * dsn);
        wv = (a * e + 1e-10f) * uls[tx * BB + b];
    }
    wsh[tx] = wv; red[tx] = wv;
    __syncthreads();
    for (int s = 64; s > 0; s >>= 1) { if (tx < s) red[tx] += red[tx + s]; __syncthreads(); }
    float sw = red[0];
    float acc = 0.f;
    for (int j = 0; j <= i; j++)
        acc = fmaf(wsh[j], rout[(size_t)(j * BB + b) * HH + tx], acc);
    size_t m = (size_t)b * LL + i;             // output row order = [B,L]
    A16[m * 256 + tx]       = (_Float16)(acc / sw);
    A16[m * 256 + 128 + tx] = (_Float16)user_emb[(size_t)uid[b] * HH + tx];
}

// ---------------- FC GEMM: Y[6400,20000] = A16[6400,256] x fc_W^T + fc_b ----------------
// block = 8 waves; block tile 64(M) x 256(N); wave tile 64x32 = 4x2 WMMA tiles
// A16/B16 are L2-resident and reused; Y is a 512MB write-once stream -> NT stores.
__global__ void __launch_bounds__(256) k_gemm(const _Float16* __restrict__ A,
        const _Float16* __restrict__ Bm, const float* __restrict__ bias,
        float* __restrict__ Y) {
    int lane = threadIdx.x & 31, wave = threadIdx.x >> 5;
    int lrow = lane & 15, lhi = lane >> 4;
    int m0 = blockIdx.y * 64;
    int n0 = blockIdx.x * 256 + wave * 32;
    if (n0 >= PP) return;                       // wave-uniform guard (20000 % 32 == 0)
    v8f acc[4][2] = {};
    for (int kb = 0; kb < 256; kb += 32) {
        Frag a[4], bf[2];
#pragma unroll
        for (int mi = 0; mi < 4; mi++) {
            const _Float16* ap = A + (size_t)(m0 + mi * 16 + lrow) * 256 + kb + lhi * 16;
            a[mi].u[0] = *(const u32x4*)ap;
            a[mi].u[1] = *(const u32x4*)(ap + 8);
        }
#pragma unroll
        for (int ni = 0; ni < 2; ni++) {
            const _Float16* bp = Bm + (size_t)(n0 + ni * 16 + lrow) * 256 + kb + lhi * 16;
            bf[ni].u[0] = *(const u32x4*)bp;
            bf[ni].u[1] = *(const u32x4*)(bp + 8);
        }
#pragma unroll
        for (int mi = 0; mi < 4; mi++)
#pragma unroll
            for (int ni = 0; ni < 2; ni++)
                acc[mi][ni] = __builtin_amdgcn_wmma_f32_16x16x32_f16(
                    false, a[mi].h, false, bf[ni].h, (short)0, acc[mi][ni], false, false);
    }
#pragma unroll
    for (int ni = 0; ni < 2; ni++) {
        int n = n0 + ni * 16 + lrow;
        float bv = bias[n];
#pragma unroll
        for (int mi = 0; mi < 4; mi++) {
#pragma unroll
            for (int r = 0; r < 8; r++) {
                int m = m0 + mi * 16 + r + lhi * 8;
                __builtin_nontemporal_store(acc[mi][ni][r] + bv, &Y[(size_t)m * PP + n]);
            }
        }
    }
}

extern "C" void kernel_launch(void* const* d_in, const int* in_sizes, int n_in,
                              void* d_out, int out_size, void* d_ws, size_t ws_size,
                              hipStream_t stream) {
    const int*   POI      = (const int*)d_in[0];
    const int*   ts       = (const int*)d_in[1];
    const float* lat      = (const float*)d_in[2];
    const float* lon      = (const float*)d_in[3];
    const int*   uid      = (const int*)d_in[4];
    const int*   grow     = (const int*)d_in[5];
    const int*   gcol     = (const int*)d_in[6];
    const float* gval     = (const float*)d_in[7];
    const int*   irow     = (const int*)d_in[8];
    const int*   icol     = (const int*)d_in[9];
    const float* ival     = (const float*)d_in[10];
    const float* enc_emb  = (const float*)d_in[11];
    const float* user_emb = (const float*)d_in[12];
    const float* memory   = (const float*)d_in[13];
    const float* W_ih     = (const float*)d_in[14];
    const float* W_hh     = (const float*)d_in[15];
    const float* b_ih     = (const float*)d_in[16];
    const float* b_hh     = (const float*)d_in[17];
    const float* fc_W     = (const float*)d_in[18];
    const float* fc_b     = (const float*)d_in[19];
    float* Y = (float*)d_out;

    // workspace layout (~37 MB total)
    float* enc_w  = (float*)d_ws;                       // P*H
    float* enc_wu = enc_w  + (size_t)PP * HH;           // U*H
    float* x_emb  = enc_wu + (size_t)UU * HH;           // L*B*H
    float* uls    = x_emb  + (size_t)LL * BB * HH;      // L*B
    float* X      = uls    + (size_t)LL * BB;           // L*B*H
    float* rout   = X      + (size_t)LL * BB * HH;      // L*B*H
    _Float16* A16 = (_Float16*)(rout + (size_t)LL * BB * HH); // 6400*256 f16
    _Float16* B16 = A16 + (size_t)LL * BB * 2 * HH;           // 20000*256 f16

    int E1 = in_sizes[5], E2 = in_sizes[8];

    int nz = (PP + UU) * HH;
    k_zero<<<(nz + 255) / 256, 256, 0, stream>>>(enc_w, nz);
    k_cvt<<<(PP * 2 * HH + 255) / 256, 256, 0, stream>>>(fc_W, B16, PP * 2 * HH);
    k_scatter<<<E1, HH, 0, stream>>>(grow, gcol, gval, enc_emb, enc_w);
    k_scatter<<<E2, HH, 0, stream>>>(irow, icol, ival, enc_emb, enc_wu);
    k_gather_uls<<<LL * BB, HH, 0, stream>>>(POI, uid, enc_w, enc_wu, x_emb, uls);
    k_rnn_input<<<LL * BB, HH, 0, stream>>>(x_emb, W_ih, b_ih, b_hh, X);
    k_rnn<<<1, 1024, 0, stream>>>(X, W_hh, memory, uid, rout);
    k_flash<<<LL * BB, HH, 0, stream>>>(ts, lat, lon, uls, rout, user_emb, uid, A16);
    k_gemm<<<dim3((PP + 255) / 256, (LL * BB) / 64), 256, 0, stream>>>(A16, B16, fc_b, Y);
}
